// Model_75582834475294
// MI455X (gfx1250) — compile-verified, matrix operand order checked by reading, not run on hardware
//
#include <hip/hip_runtime.h>

#define HH 512
#define G3 1536
#define BB 32
#define SS 392
#define MROWS (SS * BB)   // 12544
#define VV 30000
#define TT 20
#define DR (TT * BB)      // 640
#define NLAYER 8

typedef __attribute__((ext_vector_type(16))) __bf16 v16bf;
typedef __attribute__((ext_vector_type(16))) unsigned short v16u;
typedef __attribute__((ext_vector_type(8)))  float v8f;
typedef __attribute__((ext_vector_type(4)))  unsigned int u32x4;
typedef __attribute__((ext_vector_type(4)))  int i32x4;
typedef __attribute__((ext_vector_type(8)))  int i32x8;

__device__ __forceinline__ unsigned short f2bf(float f) {
  unsigned u = __float_as_uint(f);
  u += 0x7fffu + ((u >> 16) & 1u);
  return (unsigned short)(u >> 16);
}
__device__ __forceinline__ unsigned pk2bf(float lo, float hi) {
  return (unsigned)f2bf(lo) | ((unsigned)f2bf(hi) << 16);
}

// A/B fragment per CDNA5 16-bit WMMA layout: element j -> k = j + 8*half + 8*(j>=8),
// row/col = lane&15. Caller passes p already offset by (row*ld + kbase + half*8).
__device__ __forceinline__ v16bf frag_u16(const unsigned short* p) {
  v16u u;
#pragma unroll
  for (int j = 0; j < 8; ++j) { u[j] = p[j]; u[j + 8] = p[j + 16]; }
  return __builtin_bit_cast(v16bf, u);
}

__device__ __forceinline__ v16bf frag_f32(const float* p) {
  v16u u;
#pragma unroll
  for (int j = 0; j < 8; ++j) { u[j] = f2bf(p[j]); u[j + 8] = f2bf(p[j + 16]); }
  return __builtin_bit_cast(v16bf, u);
}

__device__ __forceinline__ v8f wmma_bf16(v16bf a, v16bf b, v8f c) {
  return __builtin_amdgcn_wmma_f32_16x16x32_bf16(false, a, false, b, (short)0, c,
                                                 false, false);
}

// ---------------------------------------------------------------------------
// Tensor Data Mover: DMA one bf16 tile [rows x cols], row stride = stride_elts,
// from global into LDS (contiguous rows, cols*2 bytes per row).
// Issue from one wave only (TDM ignores EXEC); caller syncs the block after.
// ---------------------------------------------------------------------------
__device__ __forceinline__ void tdm_load_bf16_tile(const unsigned short* gptr,
                                                   unsigned lds_byte_off,
                                                   unsigned rows, unsigned cols,
                                                   unsigned stride_elts) {
  unsigned long long ga = (unsigned long long)(size_t)gptr;
  u32x4 g0;
  g0[0] = 1u;                                            // count=1, user mode
  g0[1] = lds_byte_off;                                  // lds_addr
  g0[2] = (unsigned)(ga & 0xffffffffu);                  // global_addr lo
  g0[3] = (unsigned)((ga >> 32) & 0x01ffffffu) | (2u << 30);  // addr hi | type=2
  i32x8 g1;
  g1[0] = (int)(1u << 16);                 // data_size = 2 bytes
  g1[1] = (int)((cols & 0xffffu) << 16);   // tensor_dim0[15:0] @ bit48
  g1[2] = (int)((rows & 0xffffu) << 16);   // tensor_dim1[15:0] @ bit80
  g1[3] = (int)((cols & 0xffffu) << 16);   // tile_dim0 @ bit112
  g1[4] = (int)rows;                       // tile_dim1 @ bit128
  g1[5] = (int)stride_elts;                // tensor_dim0_stride[31:0] @ bit160
  g1[6] = 0;
  g1[7] = 0;
  i32x4 z4 = {0, 0, 0, 0};
#if __has_include(<hip/amd_detail/amd_gfx1250_TDM.h>)
  i32x8 z8 = {0, 0, 0, 0, 0, 0, 0, 0};
  __builtin_amdgcn_tensor_load_to_lds(g0, g1, z4, z4, z8, 0);
#else
  __builtin_amdgcn_tensor_load_to_lds(g0, g1, z4, z4, 0);
#endif
}

__device__ __forceinline__ float block_sum(float v, float* red) {
  int tid = threadIdx.x;
  red[tid] = v; __syncthreads();
  for (int s = 128; s > 0; s >>= 1) {
    if (tid < s) red[tid] += red[tid + s];
    __syncthreads();
  }
  float r = red[0]; __syncthreads();
  return r;
}

__device__ __forceinline__ float sigm(float x) { return 1.f / (1.f + expf(-x)); }

// ---------------------------------------------------------------------------
// Encoder: x = LN(img_row @ enc_W^T + enc_b) -> bf16 seq (time-major rows t*32+b)
// ---------------------------------------------------------------------------
__global__ void encoder_kernel(const float* __restrict__ Img1,
                               const float* __restrict__ Img2,
                               const float* __restrict__ encW,
                               const float* __restrict__ encb,
                               const float* __restrict__ a2,
                               const float* __restrict__ b2,
                               unsigned short* __restrict__ seq) {
  __shared__ float xs[1024];
  __shared__ float ys[HH];
  __shared__ float red[256];
  const int row = blockIdx.x;
  const int t = row >> 5, b = row & 31;
  const int tid = threadIdx.x;
  const float* img = (t < 196) ? Img1 : Img2;
  const int sp = (t < 196) ? t : (t - 196);
  for (int ci = tid; ci < 1024; ci += 256)
    xs[ci] = img[(size_t)(b * 1024 + ci) * 196 + sp];
  __syncthreads();
  for (int o = tid; o < HH; o += 256) {
    const float* w = encW + (size_t)o * 1024;
    float s = encb[o];
    for (int ci = 0; ci < 1024; ci += 4)
      s += xs[ci] * w[ci] + xs[ci + 1] * w[ci + 1] + xs[ci + 2] * w[ci + 2] +
           xs[ci + 3] * w[ci + 3];
    ys[o] = s;
  }
  __syncthreads();
  float ls = 0.f;
  for (int o = tid; o < HH; o += 256) ls += ys[o];
  float mean = block_sum(ls, red) / (float)HH;
  float ld = 0.f;
  for (int o = tid; o < HH; o += 256) { float d = ys[o] - mean; ld += d * d; }
  float denom = sqrtf(block_sum(ld, red) / (float)(HH - 1)) + 1e-6f;
  for (int o = tid; o < HH; o += 256)
    seq[(size_t)row * HH + o] = f2bf(a2[o] * (ys[o] - mean) / denom + b2[o]);
}

// ---------------------------------------------------------------------------
// Generic WMMA GEMM: C[M x N] = A_bf16[M x K] @ W_f32[N x K]^T + bias
// Block tile 64x64, 8 waves, 2 C tiles/wave. A panel (64 x K bf16) DMA'd into
// LDS once via the Tensor Data Mover; B staged per k-step with f32->bf16 cvt.
// M multiple of 64; N guarded only when GUARD (logits tail).
// ---------------------------------------------------------------------------
template <bool GUARD>
__global__ void gemm_bf16w_kernel(const unsigned short* __restrict__ A, int lda,
                                  const float* __restrict__ W,
                                  const float* __restrict__ bias,
                                  float* __restrict__ C, int ldc, int N, int K) {
  extern __shared__ unsigned char gsm[];
  unsigned short* At = (unsigned short*)gsm;               // [64][K]
  unsigned short* Bt = (unsigned short*)(gsm + (size_t)64 * K * 2);  // [64][32]
  unsigned* Bt32 = (unsigned*)Bt;
  const int tid = threadIdx.x;
  const int m0 = blockIdx.x * 64;
  const int n0 = blockIdx.y * 64;
  const int lane = tid & 31, wave = tid >> 5, half = lane >> 4, mn = lane & 15;
  const int ntl = wave & 3, mta = wave >> 2;
  // one-shot TDM load of the whole A panel for this block
  if (wave == 0) {
    tdm_load_bf16_tile(A + (size_t)m0 * lda, (unsigned)(size_t)At, 64,
                       (unsigned)K, (unsigned)lda);
    __builtin_amdgcn_s_wait_tensorcnt(0);
  }
  __syncthreads();
  v8f acc0 = {}; v8f acc1 = {};
  const int sr = tid >> 2;           // staging row 0..63
  const int sc = (tid & 3) * 8;      // staging col base
  const int nstage = n0 + sr;
  const bool ok = !GUARD || (nstage < N);
  const float* wrow = W + (size_t)nstage * K + sc;
  const unsigned short* apx0 = At + (size_t)(mta * 16 + mn) * K + half * 8;
  const unsigned short* apx1 = At + (size_t)((mta + 2) * 16 + mn) * K + half * 8;
  for (int ks = 0; ks < K / 32; ++ks) {
#pragma unroll
    for (int j = 0; j < 4; ++j) {
      float f0 = ok ? wrow[ks * 32 + 2 * j] : 0.f;
      float f1 = ok ? wrow[ks * 32 + 2 * j + 1] : 0.f;
      Bt32[(sr * 32 + sc) / 2 + j] = pk2bf(f0, f1);
    }
    __syncthreads();
    v16bf bf = frag_u16(Bt + (ntl * 16 + mn) * 32 + half * 8);
    acc0 = wmma_bf16(frag_u16(apx0 + ks * 32), bf, acc0);
    acc1 = wmma_bf16(frag_u16(apx1 + ks * 32), bf, acc1);
    __syncthreads();
  }
  const int n = n0 + ntl * 16 + mn;
  if (!GUARD || n < N) {
    const float bv = bias ? bias[n] : 0.f;
#pragma unroll
    for (int e = 0; e < 8; ++e) {
      int m = m0 + mta * 16 + e + 8 * half;
      C[(size_t)m * ldc + n] = acc0[e] + bv;
      m = m0 + (mta + 2) * 16 + e + 8 * half;
      C[(size_t)m * ldc + n] = acc1[e] + bv;
    }
  }
}

// ---------------------------------------------------------------------------
// Persistent GRU scan: 8 blocks, each owns 64 hidden units (all 3 gates).
// Whh chunk staged once in LDS (bf16). Global spin barrier between steps.
// ---------------------------------------------------------------------------
__global__ void gru_scan_kernel(const float* __restrict__ GX,   // [MROWS][1536]
                                const float* __restrict__ Whh,  // [1536][512]
                                const float* __restrict__ bhh,  // [1536]
                                float* __restrict__ Hpp,        // [2][32][512]
                                unsigned short* __restrict__ seq_out,
                                float* __restrict__ Gout,       // [32][512]
                                unsigned int* __restrict__ cnt) {
  extern __shared__ unsigned char smem[];
  unsigned short* Wl = (unsigned short*)smem;   // [192][512] bf16
  unsigned short* hb = Wl + 192 * HH;           // [32][512] bf16
  float* bl = (float*)(hb + BB * HH);           // [192]
  unsigned* Wl32 = (unsigned*)Wl;
  unsigned* hb32 = (unsigned*)hb;
  const int tid = threadIdx.x;
  const int j0 = blockIdx.x * 64;
  for (int i = tid; i < 192 * HH / 2; i += 256) {
    int g = (2 * i) >> 9, k = (2 * i) & (HH - 1);
    int gr = (g >> 6) * HH + j0 + (g & 63);     // gate*512 + j
    const float* wr = Whh + (size_t)gr * HH + k;
    Wl32[i] = pk2bf(wr[0], wr[1]);
  }
  for (int idx = tid; idx < 192; idx += 256)
    bl[idx] = bhh[(idx >> 6) * HH + j0 + (idx & 63)];
  __syncthreads();
  const int lane = tid & 31, wave = tid >> 5, half = lane >> 4, mn = lane & 15;
  const int ntl = wave & 3, mt = wave >> 2;   // 2 m-tiles x 4 n-tiles = 8 waves
  const unsigned short* ap  = hb + (size_t)(mt * 16 + mn) * HH + half * 8;
  const unsigned short* bpr = Wl + (size_t)(ntl * 16 + mn) * HH + half * 8;
  const unsigned short* bpz = bpr + (size_t)64 * HH;
  const unsigned short* bpn = bpr + (size_t)128 * HH;
  for (int t = 0; t < SS; ++t) {
    if (t > 0) {
      if (tid == 0) {
        const unsigned target = (unsigned)t * gridDim.x;
        while (atomicAdd(cnt, 0u) < target) __builtin_amdgcn_s_sleep(2);
      }
      __syncthreads();
      __threadfence();
    }
    const float* hin = Hpp + (size_t)(t & 1) * (BB * HH);
    float* hout = Hpp + (size_t)((t + 1) & 1) * (BB * HH);
    for (int i = tid; i < BB * HH / 2; i += 256)
      hb32[i] = pk2bf(hin[2 * i], hin[2 * i + 1]);
    __syncthreads();
    v8f ar = {}; v8f az = {}; v8f an = {};
#pragma unroll
    for (int ks = 0; ks < HH / 32; ++ks) {
      v16bf a = frag_u16(ap + ks * 32);
      ar = wmma_bf16(a, frag_u16(bpr + ks * 32), ar);
      az = wmma_bf16(a, frag_u16(bpz + ks * 32), az);
      an = wmma_bf16(a, frag_u16(bpn + ks * 32), an);
    }
#pragma unroll
    for (int e = 0; e < 8; ++e) {
      const int m = mt * 16 + e + 8 * half;
      const int jl = ntl * 16 + mn;
      const int j = j0 + jl;
      const float* gx = GX + (size_t)(t * BB + m) * G3;
      float r = sigm(gx[j] + ar[e] + bl[jl]);
      float z = sigm(gx[HH + j] + az[e] + bl[64 + jl]);
      float n = tanhf(gx[2 * HH + j] + r * (an[e] + bl[128 + jl]));
      float hold = hin[m * HH + j];
      float hnew = (1.f - z) * n + z * hold;
      hout[m * HH + j] = hnew;
      seq_out[(size_t)(t * BB + m) * HH + j] = f2bf(hnew);
      if (t == SS - 1) Gout[m * HH + j] = hnew;
    }
    __threadfence();
    __syncthreads();
    if (tid == 0) atomicAdd(cnt, 1u);
  }
}

// ---------------------------------------------------------------------------
// Decoder GRU cell (no biases): 8 blocks x 64 hidden units, all 3 gates/wave.
// ---------------------------------------------------------------------------
__global__ void dec_cell_kernel(const unsigned short* __restrict__ Xin,
                                const float* __restrict__ Hin,
                                const float* __restrict__ Wih,
                                const float* __restrict__ Whh,
                                float* __restrict__ Hout,
                                unsigned short* __restrict__ Xout) {
  extern __shared__ unsigned char smem[];
  unsigned short* xb = (unsigned short*)smem;   // [32][512]
  unsigned short* hb = xb + BB * HH;            // [32][512]
  const int tid = threadIdx.x;
  const int j0 = blockIdx.x * 64;
  for (int i = tid; i < BB * HH / 2; i += 256) {
    ((unsigned*)xb)[i] = ((const unsigned*)Xin)[i];
    ((unsigned*)hb)[i] = pk2bf(Hin[2 * i], Hin[2 * i + 1]);
  }
  __syncthreads();
  const int lane = tid & 31, wave = tid >> 5, half = lane >> 4, mn = lane & 15;
  const int ntl = wave & 3, mt = wave >> 2;
  const unsigned short* axp = xb + (size_t)(mt * 16 + mn) * HH + half * 8;
  const unsigned short* ahp = hb + (size_t)(mt * 16 + mn) * HH + half * 8;
  const int nr = j0 + ntl * 16 + mn;
  const float* wir = Wih + (size_t)nr * HH + half * 8;
  const float* wiz = wir + (size_t)HH * HH;
  const float* win = wir + (size_t)2 * HH * HH;
  const float* whr = Whh + (size_t)nr * HH + half * 8;
  const float* whz = whr + (size_t)HH * HH;
  const float* whn = whr + (size_t)2 * HH * HH;
  v8f ar = {}; v8f az = {}; v8f anx = {}; v8f anh = {};
  for (int ks = 0; ks < HH / 32; ++ks) {
    if (ks + 1 < HH / 32) {   // L2-resident weight rows: hint next k-slab
      __builtin_prefetch(wir + (ks + 1) * 32, 0, 3);
      __builtin_prefetch(whr + (ks + 1) * 32, 0, 3);
    }
    v16bf ax = frag_u16(axp + ks * 32);
    v16bf ah = frag_u16(ahp + ks * 32);
    ar  = wmma_bf16(ax, frag_f32(wir + ks * 32), ar);
    ar  = wmma_bf16(ah, frag_f32(whr + ks * 32), ar);
    az  = wmma_bf16(ax, frag_f32(wiz + ks * 32), az);
    az  = wmma_bf16(ah, frag_f32(whz + ks * 32), az);
    anx = wmma_bf16(ax, frag_f32(win + ks * 32), anx);
    anh = wmma_bf16(ah, frag_f32(whn + ks * 32), anh);
  }
#pragma unroll
  for (int e = 0; e < 8; ++e) {
    const int m = mt * 16 + e + 8 * half;
    const int j = j0 + ntl * 16 + mn;
    float r = sigm(ar[e]);
    float z = sigm(az[e]);
    float n = tanhf(anx[e] + r * anh[e]);
    float hold = Hin[m * HH + j];
    float hnew = (1.f - z) * n + z * hold;
    Hout[m * HH + j] = hnew;
    Xout[m * HH + j] = f2bf(hnew);
  }
}

// ---------------------------------------------------------------------------
// h0n = tanh(cat(h0_old, h8) @ dec_linW^T); writes new h[0] and decoder output
// ---------------------------------------------------------------------------
__global__ void dec_h0_kernel(const float* __restrict__ H0old,
                              const float* __restrict__ H8,
                              const float* __restrict__ Wlin,  // [512][1024]
                              float* __restrict__ H0new,
                              float* __restrict__ Orow) {
  extern __shared__ unsigned char smem[];
  unsigned short* ab = (unsigned short*)smem;   // [32][1024]
  unsigned* ab32 = (unsigned*)ab;
  const int tid = threadIdx.x;
  const int n0 = blockIdx.x * 64;
  for (int i = tid; i < BB * HH / 2; i += 256) {
    int m = (2 * i) >> 9, k = (2 * i) & (HH - 1);
    ab32[(m * 1024 + k) / 2] = pk2bf(H0old[2 * i], H0old[2 * i + 1]);
    ab32[(m * 1024 + HH + k) / 2] = pk2bf(H8[2 * i], H8[2 * i + 1]);
  }
  __syncthreads();
  const int lane = tid & 31, wave = tid >> 5, half = lane >> 4, mn = lane & 15;
  const int ntl = wave & 3, mt = wave >> 2;
  const unsigned short* apx = ab + (size_t)(mt * 16 + mn) * 1024 + half * 8;
  const int nr = n0 + ntl * 16 + mn;
  const float* wp = Wlin + (size_t)nr * 1024 + half * 8;
  v8f acc = {};
  for (int ks = 0; ks < 1024 / 32; ++ks)
    acc = wmma_bf16(frag_u16(apx + ks * 32), frag_f32(wp + ks * 32), acc);
#pragma unroll
  for (int e = 0; e < 8; ++e) {
    const int m = mt * 16 + e + 8 * half;
    const int n = n0 + ntl * 16 + mn;
    float v = tanhf(acc[e]);
    H0new[m * HH + n] = v;
    Orow[m * HH + n] = v;
  }
}

// ---------------------------------------------------------------------------
// Embedding gather -> bf16 (time-major rows t*32+b), tokens Des[b][t], t<20
// ---------------------------------------------------------------------------
__global__ void gather_kernel(const int* __restrict__ Des,
                              const float* __restrict__ emb,
                              unsigned short* __restrict__ XD) {
  const int i = blockIdx.x * 256 + threadIdx.x;
  if (i >= DR * HH) return;
  const int r = i >> 9, c = i & (HH - 1);
  const int t = r >> 5, b = r & 31;
  const int tok = Des[b * 21 + t];
  XD[i] = f2bf(emb[(size_t)tok * HH + c]);
}

// LayerNorm rows of decoder output -> bf16
__global__ void ln_out_kernel(const float* __restrict__ X,
                              const float* __restrict__ a2,
                              const float* __restrict__ b2,
                              unsigned short* __restrict__ Y) {
  __shared__ float red[256];
  const int row = blockIdx.x, tid = threadIdx.x;
  const float* x = X + (size_t)row * HH;
  float ls = 0.f;
  for (int o = tid; o < HH; o += 256) ls += x[o];
  float mean = block_sum(ls, red) / (float)HH;
  float ld = 0.f;
  for (int o = tid; o < HH; o += 256) { float d = x[o] - mean; ld += d * d; }
  float denom = sqrtf(block_sum(ld, red) / (float)(HH - 1)) + 1e-6f;
  for (int o = tid; o < HH; o += 256)
    Y[(size_t)row * HH + o] = f2bf(a2[o] * (x[o] - mean) / denom + b2[o]);
}

// log-softmax + NLL per logits row
__global__ void nll_kernel(const float* __restrict__ LG,
                           const int* __restrict__ Des,
                           float* __restrict__ out) {
  __shared__ float red[256];
  const int row = blockIdx.x, tid = threadIdx.x;
  const int t = row >> 5, b = row & 31;
  const float* x = LG + (size_t)row * VV;
  float mx = -1e30f;
  for (int i = tid; i < VV; i += 256) mx = fmaxf(mx, x[i]);
  red[tid] = mx; __syncthreads();
  for (int s = 128; s > 0; s >>= 1) {
    if (tid < s) red[tid] = fmaxf(red[tid], red[tid + s]);
    __syncthreads();
  }
  mx = red[0]; __syncthreads();
  float sm = 0.f;
  for (int i = tid; i < VV; i += 256) sm += expf(x[i] - mx);
  sm = block_sum(sm, red);
  if (tid == 0) {
    const int tgt = Des[b * 21 + (t + 1)];
    const float nll = -(x[tgt] - mx - logf(sm));
    out[row] = (tgt == 0) ? 0.f : nll;
  }
}

// ---------------------------------------------------------------------------
extern "C" void kernel_launch(void* const* d_in, const int* in_sizes, int n_in,
                              void* d_out, int out_size, void* d_ws,
                              size_t ws_size, hipStream_t stream) {
  (void)in_sizes; (void)n_in; (void)out_size; (void)ws_size;
  const float* Img1   = (const float*)d_in[0];
  const float* Img2   = (const float*)d_in[1];
  const int*   Des    = (const int*)d_in[2];
  const float* emb    = (const float*)d_in[3];
  const float* encW   = (const float*)d_in[4];
  const float* encb   = (const float*)d_in[5];
  const float* enc_a2 = (const float*)d_in[6];
  const float* enc_b2 = (const float*)d_in[7];
  const float* gWih   = (const float*)d_in[8];
  const float* gWhh   = (const float*)d_in[9];
  const float* gbih   = (const float*)d_in[10];
  const float* gbhh   = (const float*)d_in[11];
  const float* dWih   = (const float*)d_in[12];
  const float* dWhh   = (const float*)d_in[13];
  const float* dlinW  = (const float*)d_in[14];
  const float* dec_a2 = (const float*)d_in[15];
  const float* dec_b2 = (const float*)d_in[16];
  const float* outW   = (const float*)d_in[17];
  const float* outb   = (const float*)d_in[18];
  float* out = (float*)d_out;

  char* ws = (char*)d_ws;
  size_t off = 0;
  auto alloc = [&](size_t bytes) {
    size_t o = off;
    off = (off + bytes + 255) & ~(size_t)255;
    return o;
  };
  unsigned short* SEQ0 = (unsigned short*)(ws + alloc((size_t)MROWS * HH * 2));
  unsigned short* SEQ1 = (unsigned short*)(ws + alloc((size_t)MROWS * HH * 2));
  float* GXp = (float*)(ws + alloc((size_t)MROWS * G3 * 4));  // reused as logits
  float* HPp = (float*)(ws + alloc((size_t)2 * BB * HH * 4));
  float* GBp = (float*)(ws + alloc((size_t)NLAYER * BB * HH * 4));
  float* HD0 = (float*)(ws + alloc((size_t)NLAYER * BB * HH * 4));
  float* HD1 = (float*)(ws + alloc((size_t)NLAYER * BB * HH * 4));
  unsigned short* XDp = (unsigned short*)(ws + alloc((size_t)DR * HH * 2));
  unsigned short* XC0 = (unsigned short*)(ws + alloc((size_t)BB * HH * 2));
  unsigned short* XC1 = (unsigned short*)(ws + alloc((size_t)BB * HH * 2));
  float* DOp = (float*)(ws + alloc((size_t)DR * HH * 4));
  unsigned short* DOB = (unsigned short*)(ws + alloc((size_t)DR * HH * 2));
  unsigned int* CNTp = (unsigned int*)(ws + alloc(256));

  const size_t GEMM_LDS = (size_t)64 * HH * 2 + (size_t)64 * 32 * 2;  // A panel + B tile
  const size_t SCAN_LDS = (size_t)(192 * HH + BB * HH) * 2 + 192 * 4;
  const size_t DEC_LDS = (size_t)2 * BB * HH * 2;
  const size_t H0_LDS = (size_t)BB * 1024 * 2;

  hipMemsetAsync(CNTp, 0, 256, stream);

  // Encoder linear + layernorm
  encoder_kernel<<<MROWS, 256, 0, stream>>>(Img1, Img2, encW, encb, enc_a2,
                                            enc_b2, SEQ0);

  // Encoder GRU stack
  unsigned short* sin = SEQ0;
  unsigned short* sout = SEQ1;
  for (int l = 0; l < NLAYER; ++l) {
    gemm_bf16w_kernel<false><<<dim3(MROWS / 64, G3 / 64), 256, GEMM_LDS, stream>>>(
        sin, HH, gWih + (size_t)l * G3 * HH, gbih + (size_t)l * G3, GXp, G3, G3,
        HH);
    hipMemsetAsync(HPp, 0, (size_t)2 * BB * HH * 4, stream);
    gru_scan_kernel<<<8, 256, SCAN_LDS, stream>>>(
        GXp, gWhh + (size_t)l * G3 * HH, gbhh + (size_t)l * G3, HPp, sout,
        GBp + (size_t)l * BB * HH, CNTp + l);
    unsigned short* tmp = sin; sin = sout; sout = tmp;
  }

  // Decoder
  gather_kernel<<<(DR * HH + 255) / 256, 256, 0, stream>>>(Des, emb, XDp);
  hipMemcpyAsync(HD0, GBp, (size_t)NLAYER * BB * HH * 4,
                 hipMemcpyDeviceToDevice, stream);
  float* hds[2] = {HD0, HD1};
  for (int t = 0; t < TT; ++t) {
    const float* hin = hds[t & 1];
    float* hout = hds[(t + 1) & 1];
    const unsigned short* xin = XDp + (size_t)t * BB * HH;
    for (int i = 0; i < NLAYER; ++i) {
      unsigned short* xo = (i & 1) ? XC1 : XC0;
      dec_cell_kernel<<<8, 256, DEC_LDS, stream>>>(
          xin, hin + (size_t)i * BB * HH, dWih + (size_t)i * G3 * HH,
          dWhh + (size_t)i * G3 * HH, hout + (size_t)i * BB * HH, xo);
      xin = xo;
    }
    dec_h0_kernel<<<8, 256, H0_LDS, stream>>>(
        hin /* h[0] old */, hout + (size_t)7 * BB * HH /* h8 new */, dlinW,
        hout /* h[0] new */, DOp + (size_t)t * BB * HH);
  }

  // Output head
  ln_out_kernel<<<DR, 256, 0, stream>>>(DOp, dec_a2, dec_b2, DOB);
  gemm_bf16w_kernel<true><<<dim3(DR / 64, (VV + 63) / 64), 256, GEMM_LDS, stream>>>(
      DOB, HH, outW, outb, GXp, VV, VV, HH);
  nll_kernel<<<DR, 256, 0, stream>>>(GXp, Des, out);
}
